// ContextualAttention_57320633532552
// MI455X (gfx1250) — matrix operand order repaired and checked by reading, not compiled, not run on hardware
//
#include <hip/hip_runtime.h>

// ContextualAttention (DeepFill-style) for MI455X / gfx1250, wave32 + WMMA bf16.
//
// Pipeline (B=4, C=96, H=W=128, rate=2 -> 64x64 working res):
//  1) downsample f,b (nearest, "align-corners" rounding)
//  2) build A  = L2-normalized 3x3 patches of b_down  -> bf16 [s][p][864]
//     build Fc = im2col(f_down, 3x3, pad1)            -> bf16 [s][q][864]
//     build W2 = flipped 4x4 stride-2 patches of b    -> bf16 [s][su][sv][c][p]
//     build mm = (3x3 mask patch fully zero) per p
//  3) GEMM1 (WMMA bf16, f32 acc): Y[q][p] = Fc . A^T   (4096x4096x864 / sample)
//     -- double-buffered LDS, staged with GLOBAL_LOAD_ASYNC_TO_LDS_B128 (ASYNCcnt)
//  4) fuse pass1/pass2 (3x3 identity conv over flattened score map)
//  5) masked softmax over p per q -> P^T bf16 (aliases fuse ping buffer)
//  6) GEMM2 (WMMA bf16): transposed conv as 4 parity classes x 4 taps,
//     M=96(c), N=4096(pixels), K=4096(p); epilogue writes out/4.
//
// Workspace requirement: ~626 MB in d_ws (see offsets in kernel_launch).

typedef __bf16 bf16_t;
typedef __attribute__((ext_vector_type(16))) __bf16 v16bf;
typedef __attribute__((ext_vector_type(8)))  __bf16 v8bf;
typedef __attribute__((ext_vector_type(8)))  float  v8f;

#define K1 864            // 96*3*3, = 27*32
#define NP 4096           // patches / positions (64*64)
#define CCH 96
#define HW_FULL 128
#define HW_DOWN 64
#define NSAMP 4

__device__ __forceinline__ bf16_t to_bf16(float x) {
    union { float f; unsigned u; } v; v.f = x;
    unsigned r = v.u + 0x7FFFu + ((v.u >> 16) & 1u);   // round-to-nearest-even
    unsigned short h = (unsigned short)(r >> 16);
    bf16_t out;
    __builtin_memcpy(&out, &h, 2);
    return out;
}

__device__ __forceinline__ int nn_idx(int i) {
    // np.round(i * 127/63): ties never occur (254*i even vs odd), rintf is safe
    return (int)rintf((float)i * (127.0f / 63.0f));
}

// Async copy 16B/lane global -> LDS (GVS mode: SGPR64 base + VGPR 32-bit byte
// offset), tracked by ASYNCcnt.  LDS address = low 32 bits of generic pointer
__device__ __forceinline__ void async_ld_b128(unsigned lds_off, unsigned voff_bytes,
                                              unsigned long long sbase) {
    asm volatile("global_load_async_to_lds_b128 %0, %1, %2"
                 :: "v"(lds_off), "v"(voff_bytes), "s"(sbase)
                 : "memory");
}
__device__ __forceinline__ void wait_async0() {
    asm volatile("s_wait_asynccnt 0x0" ::: "memory");
}

// ---------------------------------------------------------------- downsample
__global__ void downsample_kernel(const float* __restrict__ f,
                                  const float* __restrict__ b,
                                  float* __restrict__ fd,
                                  float* __restrict__ bd) {
    int idx = blockIdx.x * blockDim.x + threadIdx.x;           // s*96*64*64
    if (idx >= NSAMP * CCH * HW_DOWN * HW_DOWN) return;
    int j = idx & 63, i = (idx >> 6) & 63;
    int c = (idx >> 12) % CCH;
    int s = idx / (CCH * HW_DOWN * HW_DOWN);
    int ih = nn_idx(i), iw = nn_idx(j);
    size_t src = (((size_t)s * CCH + c) * HW_FULL + ih) * HW_FULL + iw;
    fd[idx] = f[src];
    bd[idx] = b[src];
}

// ------------------------------------------------------------------ mm mask
__global__ void mm_kernel(const float* __restrict__ mask, float* __restrict__ mm) {
    int p = blockIdx.x * blockDim.x + threadIdx.x;
    if (p >= NP) return;
    int pi = p >> 6, pj = p & 63;
    float sum = 0.0f;
    for (int kh = 0; kh < 3; ++kh)
        for (int kw = 0; kw < 3; ++kw) {
            int ii = pi + kh - 1, jj = pj + kw - 1;
            if (ii >= 0 && ii < 64 && jj >= 0 && jj < 64)
                sum += mask[(size_t)nn_idx(ii) * HW_FULL + nn_idx(jj)];
        }
    mm[p] = (sum == 0.0f) ? 1.0f : 0.0f;
}

// ----------------------------------------------- normalized patch matrix A
__global__ __launch_bounds__(256) void build_A_kernel(const float* __restrict__ bd,
                                                      bf16_t* __restrict__ Am) {
    int gw = (blockIdx.x * blockDim.x + threadIdx.x) >> 5;     // s*4096 + p
    int lane = threadIdx.x & 31;
    if (gw >= NSAMP * NP) return;
    int s = gw >> 12, p = gw & (NP - 1);
    int pi = p >> 6, pj = p & 63;
    const float* bds = bd + (size_t)s * CCH * NP;
    float vals[27];
    float ss = 0.0f;
    #pragma unroll
    for (int t = 0; t < 27; ++t) {
        int k = t * 32 + lane;                 // k < 864 always
        int c = k / 9, rem = k % 9;
        int kh = rem / 3, kw = rem % 3;
        int ii = pi + kh - 1, jj = pj + kw - 1;
        float v = 0.0f;
        if (ii >= 0 && ii < 64 && jj >= 0 && jj < 64)
            v = bds[((size_t)c * 64 + ii) * 64 + jj];
        vals[t] = v;
        ss += v * v;
    }
    #pragma unroll
    for (int off = 16; off > 0; off >>= 1) ss += __shfl_xor(ss, off, 32);
    float inv = 1.0f / fmaxf(sqrtf(ss), 1e-4f);
    bf16_t* ar = Am + (size_t)gw * K1;
    #pragma unroll
    for (int t = 0; t < 27; ++t) ar[t * 32 + lane] = to_bf16(vals[t] * inv);
}

// ---------------------------------------------------------- im2col of f_down
__global__ void build_fc_kernel(const float* __restrict__ fd, bf16_t* __restrict__ Fc) {
    int idx = blockIdx.x * blockDim.x + threadIdx.x;           // s*4096*864
    if (idx >= NSAMP * NP * K1) return;
    int k = idx % K1;
    int q = (idx / K1) & (NP - 1);
    int s = idx / (NP * K1);
    int i = q >> 6, j = q & 63;
    int c = k / 9, rem = k % 9;
    int kh = rem / 3, kw = rem % 3;
    int ii = i + kh - 1, jj = j + kw - 1;
    float v = 0.0f;
    if (ii >= 0 && ii < 64 && jj >= 0 && jj < 64)
        v = fd[(((size_t)s * CCH + c) * 64 + ii) * 64 + jj];
    Fc[idx] = to_bf16(v);
}

// ------------------------------------- flipped raw 4x4 stride-2 patches of b
// W2[s][su][sv][c][p] = b_pad[s][c][2*pi-1+su][2*pj-1+sv]
__global__ void build_w2_kernel(const float* __restrict__ b, bf16_t* __restrict__ W2) {
    int idx = blockIdx.x * blockDim.x + threadIdx.x;           // s*4*4*96*4096
    if (idx >= NSAMP * 16 * CCH * NP) return;
    int p = idx & (NP - 1);
    int t = idx >> 12;
    int c = t % CCH; t /= CCH;
    int sv = t & 3;  t >>= 2;
    int su = t & 3;
    int s  = t >> 2;
    int pi = p >> 6, pj = p & 63;
    int row = 2 * pi - 1 + su, col = 2 * pj - 1 + sv;
    float v = 0.0f;
    if (row >= 0 && row < HW_FULL && col >= 0 && col < HW_FULL)
        v = b[(((size_t)s * CCH + c) * HW_FULL + row) * HW_FULL + col];
    W2[idx] = to_bf16(v);
}

// ------------------------------------------------- GEMM1: Y[q][p] = Fc . A^T
// block 256 = 8 waves; WG tile 128(q) x 128(p); wave tile 32(q) x 64(p).
// Double-buffered LDS, staged via async global->LDS copies.
__global__ __launch_bounds__(256) void gemm1_kernel(const bf16_t* __restrict__ Fc,
                                                    const bf16_t* __restrict__ Am,
                                                    float* __restrict__ Y) {
    __shared__ __align__(16) bf16_t As[2][128][32];   // q rows
    __shared__ __align__(16) bf16_t Bs[2][128][32];   // p rows
    const int s  = blockIdx.z;
    const int qb = blockIdx.y * 128;
    const int pb = blockIdx.x * 128;
    const int tid  = threadIdx.x;
    const int lane = tid & 31;
    const int wave = tid >> 5;
    const int wm = wave & 3;        // q-subgroup: rows wm*32 .. +31
    const int wn = wave >> 2;       // p-subgroup: cols wn*64 .. +63
    const int l16 = lane & 15, half = lane >> 4;
    const bf16_t* fc_base = Fc + (size_t)(s * NP + qb) * K1;
    const bf16_t* am_base = Am + (size_t)(s * NP + pb) * K1;
    const unsigned long long fcB = (unsigned long long)(uintptr_t)fc_base;
    const unsigned long long amB = (unsigned long long)(uintptr_t)am_base;

    // staging chunk coords (512 x 16B chunks per tile, 2 per thread per tile)
    const int row0 = tid >> 2, seg0 = tid & 3;
    const int row1 = (tid + 256) >> 2, seg1 = (tid + 256) & 3;

    v8f acc[2][4];
    #pragma unroll
    for (int i = 0; i < 2; ++i)
        #pragma unroll
        for (int j = 0; j < 4; ++j) acc[i][j] = (v8f)0.0f;

    // prologue: stage K-step 0 into buffer 0
    {
        unsigned o0 = (unsigned)(row0 * K1 + seg0 * 8) * 2u;
        unsigned o1 = (unsigned)(row1 * K1 + seg1 * 8) * 2u;
        async_ld_b128((unsigned)(uintptr_t)&As[0][row0][seg0 * 8], o0, fcB);
        async_ld_b128((unsigned)(uintptr_t)&As[0][row1][seg1 * 8], o1, fcB);
        async_ld_b128((unsigned)(uintptr_t)&Bs[0][row0][seg0 * 8], o0, amB);
        async_ld_b128((unsigned)(uintptr_t)&Bs[0][row1][seg1 * 8], o1, amB);
    }

    for (int kt = 0; kt < K1 / 32; ++kt) {
        wait_async0();          // my async writes to buf[kt&1] are done
        __syncthreads();        // everyone's writes done / last step's reads done

        if (kt + 1 < K1 / 32) { // stage next K-step into the other buffer
            int nbuf = (kt + 1) & 1;
            int k0n = (kt + 1) * 32;
            unsigned o0 = (unsigned)(row0 * K1 + k0n + seg0 * 8) * 2u;
            unsigned o1 = (unsigned)(row1 * K1 + k0n + seg1 * 8) * 2u;
            async_ld_b128((unsigned)(uintptr_t)&As[nbuf][row0][seg0 * 8], o0, fcB);
            async_ld_b128((unsigned)(uintptr_t)&As[nbuf][row1][seg1 * 8], o1, fcB);
            async_ld_b128((unsigned)(uintptr_t)&Bs[nbuf][row0][seg0 * 8], o0, amB);
            async_ld_b128((unsigned)(uintptr_t)&Bs[nbuf][row1][seg1 * 8], o1, amB);
        }

        const int buf = kt & 1;
        v16bf afrag[2], bfrag[4];
        #pragma unroll
        for (int i = 0; i < 2; ++i) {
            const bf16_t* ap = &As[buf][wm * 32 + i * 16 + l16][half * 8];
            ((v8bf*)&afrag[i])[0] = *(const v8bf*)(ap);
            ((v8bf*)&afrag[i])[1] = *(const v8bf*)(ap + 16);
        }
        #pragma unroll
        for (int j = 0; j < 4; ++j) {
            const bf16_t* bp = &Bs[buf][wn * 64 + j * 16 + l16][half * 8];
            ((v8bf*)&bfrag[j])[0] = *(const v8bf*)(bp);
            ((v8bf*)&bfrag[j])[1] = *(const v8bf*)(bp + 16);
        }
        #pragma unroll
        for (int i = 0; i < 2; ++i)
            #pragma unroll
            for (int j = 0; j < 4; ++j)
                acc[i][j] = __builtin_amdgcn_wmma_f32_16x16x32_bf16(
                    false, afrag[i], false, bfrag[j], (short)0, acc[i][j], false, false);
    }

    float* ys = Y + ((size_t)s << 24);           // s * 4096 * 4096
    #pragma unroll
    for (int i = 0; i < 2; ++i)
        #pragma unroll
        for (int j = 0; j < 4; ++j)
            #pragma unroll
            for (int r = 0; r < 8; ++r) {
                int q = qb + wm * 32 + i * 16 + r + half * 8;
                int p = pb + wn * 64 + j * 16 + l16;
                ys[((size_t)q << 12) + p] = acc[i][j][r];
            }
}

// ------------------------------------------- fuse: 3x3 identity conv, 2 modes
// mode 0: flat (q,p);  mode 1: transposed-flatten (j*64+i, bj*64+bi)
__global__ void fuse_kernel(const float* __restrict__ in, float* __restrict__ out, int mode) {
    size_t idx = (size_t)blockIdx.x * blockDim.x + threadIdx.x;
    const size_t total = (size_t)NSAMP << 24;
    if (idx >= total) return;
    size_t e = idx & 0xFFFFFFull;                // within-sample
    int q = (int)(e >> 12), p = (int)(e & 4095);
    const float* base = in + (idx - e);          // sample base
    float sum = 0.0f;
    if (mode == 0) {
        #pragma unroll
        for (int d = -1; d <= 1; ++d) {
            int r = q + d, c = p + d;
            if (r >= 0 && r < NP && c >= 0 && c < NP)
                sum += base[((size_t)r << 12) + c];
        }
    } else {
        int r2 = (q & 63) * 64 + (q >> 6);
        int c2 = (p & 63) * 64 + (p >> 6);
        #pragma unroll
        for (int d = -1; d <= 1; ++d) {
            int r = r2 + d, c = c2 + d;
            if (r >= 0 && r < NP && c >= 0 && c < NP) {
                int qq = (r & 63) * 64 + (r >> 6);
                int pp = (c & 63) * 64 + (c >> 6);
                sum += base[((size_t)qq << 12) + pp];
            }
        }
    }
    out[idx] = sum;
}

// --------------------------------- masked softmax over p per row q -> PT bf16
__global__ __launch_bounds__(256) void softmax_kernel(const float* __restrict__ Y,
                                                      const float* __restrict__ mm,
                                                      bf16_t* __restrict__ PT) {
    __shared__ float red[256];
    int row = blockIdx.x;                        // s*4096 + q
    const float* yr = Y + ((size_t)row << 12);
    bf16_t* pr = PT + ((size_t)row << 12);
    int tid = threadIdx.x;

    float mx = -1e30f;
    for (int p = tid; p < NP; p += 256)
        mx = fmaxf(mx, yr[p] * mm[p] * 10.0f);
    red[tid] = mx; __syncthreads();
    for (int off = 128; off > 0; off >>= 1) {
        if (tid < off) red[tid] = fmaxf(red[tid], red[tid + off]);
        __syncthreads();
    }
    mx = red[0]; __syncthreads();

    float sum = 0.0f;
    for (int p = tid; p < NP; p += 256)
        sum += __expf(yr[p] * mm[p] * 10.0f - mx);
    red[tid] = sum; __syncthreads();
    for (int off = 128; off > 0; off >>= 1) {
        if (tid < off) red[tid] += red[tid + off];
        __syncthreads();
    }
    float inv = 1.0f / red[0];

    for (int p = tid; p < NP; p += 256) {
        float v = __expf(yr[p] * mm[p] * 10.0f - mx) * inv * mm[p];
        pr[p] = to_bf16(v);
    }
}

// ---------------------- GEMM2: transposed conv via parity classes + WMMA bf16
// grid: (64 output rows m, 4 classes, 4 samples); block 192 = 6 waves
// out[s][c][2m+a][2n'+b] = 0.25 * sum_{di,dj,p} PT[q'(m,n',di,dj)][p] * W2[su][sv][c][p]
__global__ __launch_bounds__(192) void gemm2_kernel(const bf16_t* __restrict__ W2,
                                                    const bf16_t* __restrict__ PT,
                                                    float* __restrict__ out) {
    __shared__ __align__(16) bf16_t Bs[64][32];
    const int s   = blockIdx.z;
    const int cls = blockIdx.y;
    const int a = cls >> 1, bpar = cls & 1;
    const int m = blockIdx.x;                    // output row within class
    const int tid = threadIdx.x;
    const int lane = tid & 31;
    const int wave = tid / 32;                   // 0..5
    const int cbase = wave * 16;
    const int l16 = lane & 15, half = lane >> 4;
    const bf16_t* pts = PT + ((size_t)s << 24);

    v8f acc[4];
    #pragma unroll
    for (int j = 0; j < 4; ++j) acc[j] = (v8f)0.0f;

    for (int di = 0; di < 2; ++di) {
        int irow = m + (a ? di : di - 1);
        if (irow < 0 || irow >= 64) continue;    // uniform across block
        int u = 2 * di + a;
        for (int dj = 0; dj < 2; ++dj) {
            int v = 2 * dj + bpar;
            int su = 3 - u, sv = 3 - v;
            const bf16_t* wplane = W2 + ((((size_t)s * 4 + su) * 4 + sv) * CCH) * NP;
            int joff = bpar ? dj : dj - 1;

            for (int k0 = 0; k0 < NP; k0 += 32) {
                __syncthreads();
                for (int ck = tid; ck < 256; ck += 192) {   // 64 rows x 4 chunks
                    int n = ck >> 2, seg = ck & 3;
                    int j = n + joff;
                    uint4 val = make_uint4(0u, 0u, 0u, 0u);
                    if (j >= 0 && j < 64)
                        val = *(const uint4*)(pts + ((size_t)(irow * 64 + j) << 12)
                                              + k0 + seg * 8);
                    *(uint4*)&Bs[n][seg * 8] = val;
                }
                __syncthreads();

                v16bf afrag;
                const bf16_t* ap = wplane + ((size_t)(cbase + l16) << 12) + k0 + half * 8;
                ((v8bf*)&afrag)[0] = *(const v8bf*)(ap);
                ((v8bf*)&afrag)[1] = *(const v8bf*)(ap + 16);
                if (k0 + 32 < NP)                 // prefetch next A chunk into caches
                    __builtin_prefetch(ap + 32, 0, 0);
                #pragma unroll
                for (int jt = 0; jt < 4; ++jt) {
                    v16bf bfrag;
                    const bf16_t* bp = &Bs[jt * 16 + l16][half * 8];
                    ((v8bf*)&bfrag)[0] = *(const v8bf*)(bp);
                    ((v8bf*)&bfrag)[1] = *(const v8bf*)(bp + 16);
                    acc[jt] = __builtin_amdgcn_wmma_f32_16x16x32_bf16(
                        false, afrag, false, bfrag, (short)0, acc[jt], false, false);
                }
            }
        }
    }

    float* outs = out + (size_t)s * CCH * HW_FULL * HW_FULL;
    #pragma unroll
    for (int jt = 0; jt < 4; ++jt)
        #pragma unroll
        for (int r = 0; r < 8; ++r) {
            int c  = cbase + r + half * 8;
            int np = jt * 16 + l16;              // n' in [0,64)
            int oh = 2 * m + a, ow = 2 * np + bpar;
            outs[((size_t)c * HW_FULL + oh) * HW_FULL + ow] = acc[jt][r] * 0.25f;
        }
}

// ------------------------------------------------------------------ launcher
extern "C" void kernel_launch(void* const* d_in, const int* in_sizes, int n_in,
                              void* d_out, int out_size, void* d_ws, size_t ws_size,
                              hipStream_t stream) {
    (void)in_sizes; (void)n_in; (void)out_size; (void)ws_size;
    const float* f    = (const float*)d_in[0];   // (4,96,128,128)
    const float* b    = (const float*)d_in[1];   // (4,96,128,128)
    const float* mask = (const float*)d_in[2];   // (1,1,128,128)
    float* out = (float*)d_out;                  // (4,96,128,128)
    char* ws = (char*)d_ws;

    // workspace layout (bytes, 256-aligned); total ~626 MB
    size_t off = 0;
    auto take = [&](size_t bytes) { size_t o = off; off += (bytes + 255) & ~(size_t)255; return o; };
    float*  FD = (float*) (ws + take((size_t)NSAMP * CCH * NP * 4));            //  6.3 MB
    float*  BD = (float*) (ws + take((size_t)NSAMP * CCH * NP * 4));            //  6.3 MB
    float*  MM = (float*) (ws + take((size_t)NP * 4));                          //   16 KB
    bf16_t* AM = (bf16_t*)(ws + take((size_t)NSAMP * NP * K1 * 2));             // 28.3 MB
    bf16_t* FC = (bf16_t*)(ws + take((size_t)NSAMP * NP * K1 * 2));             // 28.3 MB
    bf16_t* W2 = (bf16_t*)(ws + take((size_t)NSAMP * 16 * CCH * NP * 2));       // 50.3 MB
    float*  Y  = (float*) (ws + take(((size_t)NSAMP << 24) * 4));               //  268 MB
    float*  YF = (float*) (ws + take(((size_t)NSAMP << 24) * 4));               //  268 MB
    bf16_t* PT = (bf16_t*)YF;   // softmax output aliases fuse ping buffer

    // 1) downsample
    {
        int n = NSAMP * CCH * HW_DOWN * HW_DOWN;
        downsample_kernel<<<(n + 255) / 256, 256, 0, stream>>>(f, b, FD, BD);
    }
    // 2) mask-validity per patch
    mm_kernel<<<(NP + 255) / 256, 256, 0, stream>>>(mask, MM);
    // 3) normalized patch matrix A (1 wave / patch)
    build_A_kernel<<<(NSAMP * NP) / 8, 256, 0, stream>>>(BD, AM);
    // 4) im2col of f_down
    {
        int n = NSAMP * NP * K1;
        build_fc_kernel<<<(n + 255) / 256, 256, 0, stream>>>(FD, FC);
    }
    // 5) flipped raw patches for the deconv
    {
        int n = NSAMP * 16 * CCH * NP;
        build_w2_kernel<<<(n + 255) / 256, 256, 0, stream>>>(b, W2);
    }
    // 6) similarity GEMM (WMMA, async-staged LDS)
    gemm1_kernel<<<dim3(NP / 128, NP / 128, NSAMP), 256, 0, stream>>>(FC, AM, Y);
    // 7) fuse passes: Y -> YF (mode 0), YF -> Y (mode 1)
    {
        size_t total = (size_t)NSAMP << 24;
        unsigned blocks = (unsigned)((total + 255) / 256);
        fuse_kernel<<<blocks, 256, 0, stream>>>(Y, YF, 0);
        fuse_kernel<<<blocks, 256, 0, stream>>>(YF, Y, 1);
    }
    // 8) masked softmax -> PT (bf16, aliases YF)
    softmax_kernel<<<NSAMP * NP, 256, 0, stream>>>(Y, MM, PT);
    // 9) transposed-conv GEMM (WMMA) -> output
    gemm2_kernel<<<dim3(HW_DOWN, 4, NSAMP), 192, 0, stream>>>(W2, PT, out);
}
